// LaplacianLoss_16569983828434
// MI455X (gfx1250) — compile-verified, matrix operand order checked by reading, not run on hardware
//
#include <hip/hip_runtime.h>

typedef __attribute__((ext_vector_type(16))) _Float16 v16h;
typedef __attribute__((ext_vector_type(8)))  float    v8f;

#define NV    8192
#define WPR   256          // 32-bit words per bitmap row
#define NCOL  96           // BATCH*3 folded output columns
#define NTILE 6            // 96/16 column tiles

// ---------------------------------------------------------------------------
// Kernel 0: zero the adjacency bitmap (8 MB) and the 32 output accumulators.
// ---------------------------------------------------------------------------
__global__ void k_zero(uint4* __restrict__ bm4, float* __restrict__ out) {
    unsigned i = blockIdx.x * 256u + threadIdx.x;
    bm4[i] = make_uint4(0u, 0u, 0u, 0u);
    if (i < 32u) out[i] = 0.0f;
}

// ---------------------------------------------------------------------------
// Kernel 1: scatter 6 directed edges per face; atomicOr dedups (ref uses .set)
// ---------------------------------------------------------------------------
__global__ void k_edges(const int* __restrict__ faces,
                        unsigned* __restrict__ bm, int nf) {
    int t = blockIdx.x * 256 + threadIdx.x;
    if (t >= nf) return;
    int v0 = faces[3 * t + 0];
    int v1 = faces[3 * t + 1];
    int v2 = faces[3 * t + 2];
    int e[6][2] = {{v0,v1},{v1,v0},{v1,v2},{v2,v1},{v2,v0},{v0,v2}};
#pragma unroll
    for (int i = 0; i < 6; ++i) {
        int v = e[i][0], w = e[i][1];
        atomicOr(&bm[(size_t)v * WPR + (w >> 5)], 1u << (w & 31));
    }
}

// ---------------------------------------------------------------------------
// Kernel 2: Xt[n][w] = (f16) x[b, w, d], n = b*3+d (w-major for b128 B loads)
// ---------------------------------------------------------------------------
__global__ void k_xt(const float* __restrict__ x, _Float16* __restrict__ xt) {
    unsigned idx = blockIdx.x * 256u + threadIdx.x;   // 96*8192 total
    unsigned w = idx & (NV - 1);
    unsigned n = idx >> 13;
    unsigned b = n / 3u, d = n - 3u * b;
    xt[idx] = (_Float16)x[((size_t)b * NV + w) * 3 + d];
}

// ---------------------------------------------------------------------------
// Kernel 3: 16-row tile of Y = L*X via v_wmma_f32_16x16x32_f16.
//  - bitmap words in registers; expanded once into 256 KB LDS A-buffer
//  - active chunks compacted into an ordered list (deterministic)
//  - 4-slot software pipeline, prefetch distance 3, hides L2 latency
// ---------------------------------------------------------------------------
__global__ __launch_bounds__(256)
void k_spmm(const _Float16* __restrict__ xt,
            const unsigned* __restrict__ bm,
            float* __restrict__ out) {
    __shared__ uint4    Abuf[WPR * 64];          // 256 KB expanded A tiles
    __shared__ unsigned cmask[8];                // chunk activity bits
    __shared__ int      clist[WPR];              // compacted active chunks
    __shared__ int      ccount;
    __shared__ int      degcnt[16];
    __shared__ __align__(16) float rdeg[16];
    __shared__ float    bsum[32];

    const int t     = threadIdx.x;
    const int r0    = blockIdx.x << 4;           // first of 16 rows
    const int cdiag = r0 >> 5;                   // chunk holding the diagonal

    // ---- phase 1: load 16 bitmap words (rows 0..15, column-chunk = t) -----
    unsigned wrd[16];
#pragma unroll
    for (int i = 0; i < 16; ++i)
        wrd[i] = bm[(size_t)(r0 + i) * WPR + t];
    if (t < 16) degcnt[t] = 0;
    if (t < 32) bsum[t] = 0.0f;
    if (t < 8)  cmask[t] = 0u;
    __syncthreads();

    // ---- phase 2: expand chunk c = t into Abuf; degrees; activity mask ----
    {
        unsigned any = 0u;
        char* ab = (char*)Abuf + t * 1024;
        const int swz = t & 15;
#pragma unroll
        for (int i = 0; i < 16; ++i) {
            unsigned wv = wrd[i];
            any |= wv;
            uint4* dst = (uint4*)(ab + ((i ^ swz) << 4));   // bank-swizzled row
            if (wv) {
                atomicAdd(&degcnt[i], __popc(wv));
#pragma unroll
                for (int p = 0; p < 4; ++p) {               // k in [8p, 8p+8)
                    unsigned u[4];
#pragma unroll
                    for (int j = 0; j < 4; ++j) {
                        unsigned b = (wv >> (8 * p + 2 * j)) & 3u;
                        u[j] = ((b & 1u) ? 0xBC00u : 0u)        // -1.0h
                             | ((b & 2u) ? 0xBC000000u : 0u);
                    }
                    dst[p * 16] = make_uint4(u[0], u[1], u[2], u[3]);
                }
            } else {
                uint4 z = make_uint4(0u, 0u, 0u, 0u);
                dst[0] = z; dst[16] = z; dst[32] = z; dst[48] = z;
            }
        }
        if (any || t == cdiag) atomicOr(&cmask[t >> 5], 1u << (t & 31));
    }
    __syncthreads();

    // ---- phase 3a: reciprocal degree + patch diagonal (= deg) into Abuf ---
    if (t < 16) {
        int d = degcnt[t];
        rdeg[t] = (d > 0) ? 1.0f / (float)d : 0.0f;
        union { _Float16 f; unsigned short u; } cv;
        cv.f = (_Float16)(float)d;
        int dk = (r0 + t) & 31;                  // diagonal bit within chunk
        char* ab = (char*)Abuf + cdiag * 1024 + ((dk >> 3) << 8)
                 + ((t ^ (cdiag & 15)) << 4) + ((dk & 7) << 1);
        *(unsigned short*)ab = cv.u;
    }
    // ---- phase 3b: order-preserving compaction of active chunks (wave 0) --
    if (t < 32) {
        int base = 0;
#pragma unroll
        for (int cw = 0; cw < 8; ++cw) {
            unsigned cm = cmask[cw];
            if (cm & (1u << t))
                clist[base + __popc(cm & ((1u << t) - 1u))] = (cw << 5) + t;
            base += __popc(cm);
        }
        if (t == 0) ccount = base;
    }
    __syncthreads();

    // ---- phase 4: pipelined WMMA GEMM over the active-chunk list ----------
    const int wave = t >> 5, lane = t & 31;
    if (wave < NTILE) {
        const int hi = lane >> 4;                // half-wave select
        const int ml = lane & 15;                // A row / local B column
        const int n  = (wave << 4) + ml;         // global column (b*3+d)
        const uint4* xrow = (const uint4*)(xt + (size_t)n * NV);
        const int cnt = ccount;                  // >= 1 (diag chunk forced)

        union AB { uint4 q[2]; v16h h; };
        AB A0, A1, A2, A3, B0, B1, B2, B3;
        v8f acc = (v8f)0.0f;

        auto issue = [&](AB& Asl, AB& Bsl, int idx) {
            int c = clist[idx];
            const char* ab = (const char*)Abuf + c * 1024 + (hi << 8)
                           + ((ml ^ (c & 15)) << 4);
            Asl.q[0] = *(const uint4*)ab;          // k =      hi*8 .. +7
            Asl.q[1] = *(const uint4*)(ab + 512);  // k = 16 + hi*8 .. +7
            Bsl.q[0] = xrow[(c << 2) + hi];
            Bsl.q[1] = xrow[(c << 2) + hi + 2];
        };
        auto mm = [&](AB& Asl, AB& Bsl) {
            acc = __builtin_amdgcn_wmma_f32_16x16x32_f16(
                    false, Asl.h, false, Bsl.h, (short)0, acc, false, false);
        };

        if (cnt > 0) {
            issue(A0, B0, 0);
            issue(A1, B1, min(1, cnt - 1));      // dups are loaded, never MM'd
            issue(A2, B2, min(2, cnt - 1));
            issue(A3, B3, min(3, cnt - 1));
            int i = 4;
            for (; i + 3 < cnt; i += 4) {        // steady state: distance 3
                mm(A0, B0); issue(A0, B0, i);
                mm(A1, B1); issue(A1, B1, i + 1);
                mm(A2, B2); issue(A2, B2, i + 2);
                mm(A3, B3); issue(A3, B3, i + 3);
            }
            // epilogue: consume remaining slots, refill/consume tail entries
            mm(A0, B0);
            if (i     < cnt) issue(A0, B0, i);
            if (cnt > 1) { mm(A1, B1); }
            if (i + 1 < cnt) issue(A1, B1, i + 1);
            if (cnt > 2) { mm(A2, B2); }
            if (i + 2 < cnt) issue(A2, B2, i + 2);
            if (cnt > 3) { mm(A3, B3); }
            if (i     < cnt) mm(A0, B0);
            if (i + 1 < cnt) mm(A1, B1);
            if (i + 2 < cnt) mm(A2, B2);
        }

        // normalize by 1/deg, square, reduce.
        // C layout: VGPR i, lanes 0-15 -> M=i, lanes 16-31 -> M=8+i.
        float4 ra = *(const float4*)&rdeg[hi << 3];
        float4 rb = *(const float4*)&rdeg[(hi << 3) + 4];
        float r[8] = {ra.x, ra.y, ra.z, ra.w, rb.x, rb.y, rb.z, rb.w};
        float s = 0.0f;
#pragma unroll
        for (int i = 0; i < 8; ++i) {
            float y = acc[i] * r[i];
            s += y * y;
        }
        atomicAdd(&bsum[n / 3], s);
    }
    __syncthreads();
    if (t < 32) atomicAdd(&out[t], bsum[t] * (1.0f / (float)NV));
}

// ---------------------------------------------------------------------------
extern "C" void kernel_launch(void* const* d_in, const int* in_sizes, int n_in,
                              void* d_out, int out_size, void* d_ws, size_t ws_size,
                              hipStream_t stream) {
    const float* x     = (const float*)d_in[0];   // (32, 8192, 3) f32
    const int*   faces = (const int*)d_in[1];     // (16384, 3) i32
    float*       out   = (float*)d_out;           // (32,) f32

    const int nf = in_sizes[1] / 3;

    unsigned*  bm = (unsigned*)d_ws;                                    // 8 MB
    _Float16*  xt = (_Float16*)((char*)d_ws + (size_t)8 * 1024 * 1024); // 1.5 MB

    k_zero <<<2048, 256, 0, stream>>>((uint4*)bm, out);
    k_edges<<<(nf + 255) / 256, 256, 0, stream>>>(faces, bm, nf);
    k_xt   <<<(NCOL * NV) / 256, 256, 0, stream>>>(x, xt);
    k_spmm <<<NV / 16, 256, 0, stream>>>(xt, bm, out);
}
